// RBM_83786222010631
// MI455X (gfx1250) — matrix operand order repaired and checked by reading, not compiled, not run on hardware
//
#include <hip/hip_runtime.h>
#include <hip/hip_bf16.h>
#include <stdint.h>

// RBM CD-4 loss, fused WMMA implementation for gfx1250 (MI455X).
// B=16384, D=4096, H=128.  State kept in fp16 (x,h binary -> exact).
// Workspace layout (bytes):
//   [0,2MB)       Wt  : padded W^T fp16 [H+16][D]; row 128 = bx (dot-term tile), 129..143 = 0
//   [2MB,3MB)     Wh  : W fp16 [D][H]
//   [3MB,7MB)     Hs  : h_samp fp16 [B][H]
//   [8MB,136MB)   Xh  : x state fp16 [B][D]
//   [136MB,+8B)   acc : {sum F(x), sum F(x_rec)}

#define BATCH 16384
#define DIM   4096
#define HID   128

#if defined(__has_builtin)
#if __has_builtin(__builtin_amdgcn_global_load_async_to_lds_b128)
#define USE_ASYNC_LDS 1
#endif
#endif

typedef _Float16 v16h __attribute__((ext_vector_type(16)));
typedef _Float16 h8   __attribute__((ext_vector_type(8)));
typedef _Float16 h4   __attribute__((ext_vector_type(4)));
typedef float    v8f  __attribute__((ext_vector_type(8)));
typedef int      i4v  __attribute__((ext_vector_type(4)));

union F16x16 { v16h v; h8 h[2]; };

__device__ __forceinline__ float sigmoidf_(float z) {
  return 1.0f / (1.0f + __expf(-z));
}
__device__ __forceinline__ float softplusf_(float z) {
  float az = fabsf(z);
  return fmaxf(z, 0.0f) + __logf(1.0f + __expf(-az));
}
// Deterministic stateless hash RNG -> uniform [0,1)
__device__ __forceinline__ float urand_(uint32_t s, uint32_t r, uint32_t c) {
  uint32_t x = (s + 0x9E3779B9u) ^ (r * 0x85EBCA6Bu);
  x += c * 0xC2B2AE35u;
  x ^= x >> 16; x *= 0x7FEB352Du;
  x ^= x >> 15; x *= 0x846CA68Bu;
  x ^= x >> 16;
  return (float)(x >> 8) * (1.0f / 16777216.0f);
}

// 16B global->LDS copy: async DMA path (GLOBAL_LOAD_ASYNC_TO_LDS_B128, ASYNCcnt).
__device__ __forceinline__ void copy16_g2l(const _Float16* g, _Float16* l) {
#ifdef USE_ASYNC_LDS
  __builtin_amdgcn_global_load_async_to_lds_b128(
      (__attribute__((address_space(1))) i4v*)(_Float16*)g,
      (__attribute__((address_space(3))) i4v*)l, 0, 0);
#else
  *(h8*)l = *(const h8*)g;
#endif
}
__device__ __forceinline__ void wait_async_le8() {
#ifdef USE_ASYNC_LDS
#if __has_builtin(__builtin_amdgcn_s_wait_asynccnt)
  __builtin_amdgcn_s_wait_asynccnt(8);
#else
  asm volatile("s_wait_asynccnt 0x8" ::: "memory");
#endif
#endif
}
__device__ __forceinline__ void wait_async_le0() {
#ifdef USE_ASYNC_LDS
#if __has_builtin(__builtin_amdgcn_s_wait_asynccnt)
  __builtin_amdgcn_s_wait_asynccnt(0);
#else
  asm volatile("s_wait_asynccnt 0x0" ::: "memory");
#endif
#endif
}

// Forward GEMM tile: acc[16 x NT*16] = Xh[rbase:rbase+16, :] @ (padded) W  (Wt[H+16][D]).
// NT=8: plain x@W.  NT=9: also computes the bx-dot tile (Wt row 128 = bx).
template <int NT>
__device__ __forceinline__ void gemm_fwd16(const _Float16* __restrict__ Xh,
                                           const _Float16* __restrict__ Wt,
                                           int rbase, v8f (&acc)[NT]) {
  const int lane  = threadIdx.x & 31;
  const int arow  = lane & 15;
  const int akoff = (lane >> 4) << 3;   // 0 or 8
  const int bcol  = lane & 15;
  const int bkoff = (lane >> 4) << 3;   // 0 or 8
  v8f z = {};
#pragma unroll
  for (int t = 0; t < NT; ++t) acc[t] = z;
  const _Float16* aptr = Xh + (size_t)(rbase + arow) * DIM + akoff;
  for (int kc = 0; kc < DIM / 32; ++kc) {
    F16x16 a;
    a.h[0] = *(const h8*)(aptr);
    a.h[1] = *(const h8*)(aptr + 16);
    __builtin_prefetch(aptr + 4 * 32, 0, 2);   // global_prefetch_b8
#pragma unroll
    for (int t = 0; t < NT; ++t) {
      const _Float16* bp = Wt + (size_t)(t * 16 + bcol) * DIM + kc * 32 + bkoff;
      F16x16 b;
      b.h[0] = *(const h8*)(bp);
      b.h[1] = *(const h8*)(bp + 16);
      acc[t] = __builtin_amdgcn_wmma_f32_16x16x32_f16(
          false, a.v, false, b.v, (short)0, acc[t], false, false);
    }
    aptr += 32;
  }
}

// --- kernels ---------------------------------------------------------------

// W fp32 -> Wh [D][H] and padded Wt [H+16][D] fp16 (row 128 = bx, 129..143 = 0);
// zero the two accumulators.
__global__ void k_init_w(const float* __restrict__ W, const float* __restrict__ bx,
                         _Float16* __restrict__ Wh, _Float16* __restrict__ Wt,
                         float* __restrict__ acc) {
  const int d = blockIdx.x;
  const int h = threadIdx.x;
  float w = W[(size_t)d * HID + h];
  Wh[(size_t)d * HID + h] = (_Float16)w;
  Wt[(size_t)h * DIM + d] = (_Float16)w;
  if (h < 16)
    Wt[(size_t)(HID + h) * DIM + d] = (h == 0) ? (_Float16)bx[d] : (_Float16)0.0f;
  if (d == 0 && h < 2) acc[h] = 0.0f;
}

// x fp32 -> fp16 state (exact: x is binary)
__global__ void k_x2half(const float* __restrict__ x, _Float16* __restrict__ Xh) {
  const int i = blockIdx.x * blockDim.x + threadIdx.x;  // over B*D/4
  float4 v = ((const float4*)x)[i];
  h4 o = { (_Float16)v.x, (_Float16)v.y, (_Float16)v.z, (_Float16)v.w };
  ((h4*)Xh)[i] = o;
}

// Accumulate F(v) = -sum_j softplus((vW+bh)_j) - v.bx over a 16-row tile into *accp.
// Dot term comes from the 9th WMMA tile (col 0 of acc[8]).
__global__ void __launch_bounds__(128)
k_fe(const _Float16* __restrict__ Xh, const _Float16* __restrict__ Wt,
     const float* __restrict__ bh, float* __restrict__ accp) {
  const int wave  = blockIdx.x * (blockDim.x >> 5) + (threadIdx.x >> 5);
  const int rbase = wave * 16;
  const int lane  = threadIdx.x & 31;
  float bhv[8];
#pragma unroll
  for (int t = 0; t < 8; ++t) bhv[t] = bh[t * 16 + (lane & 15)];
  v8f acc[9];
  gemm_fwd16<9>(Xh, Wt, rbase, acc);
  float s = 0.0f;
#pragma unroll
  for (int t = 0; t < 8; ++t)
#pragma unroll
    for (int j = 0; j < 8; ++j)
      s += softplusf_(acc[t][j] + bhv[t]);
  float dt = 0.0f;
#pragma unroll
  for (int j = 0; j < 8; ++j) dt += acc[8][j];   // v.bx lives in col 0 of tile 8
  if ((lane & 15) == 0) s += dt;
#pragma unroll
  for (int m = 16; m >= 1; m >>= 1) s += __shfl_xor(s, m, 32);
  if (lane == 0) atomicAdd(accp, -s);
}

// h_samp = bernoulli(sigmoid(x W + bh))  -> Hs fp16 {0,1}
__global__ void __launch_bounds__(128)
k_hstep(const _Float16* __restrict__ Xh, const _Float16* __restrict__ Wt,
        const float* __restrict__ bh, _Float16* __restrict__ Hs, int step) {
  const int wave  = blockIdx.x * (blockDim.x >> 5) + (threadIdx.x >> 5);
  const int rbase = wave * 16;
  const int lane  = threadIdx.x & 31;
  float bhv[8];
#pragma unroll
  for (int t = 0; t < 8; ++t) bhv[t] = bh[t * 16 + (lane & 15)];
  v8f acc[8];
  gemm_fwd16<8>(Xh, Wt, rbase, acc);
  const int rowoff = (lane >> 4) << 3;   // C layout: hi lanes hold rows M+8
#pragma unroll
  for (int t = 0; t < 8; ++t) {
    const int col = t * 16 + (lane & 15);
#pragma unroll
    for (int j = 0; j < 8; ++j) {
      float p = sigmoidf_(acc[t][j] + bhv[t]);
      int row = rbase + j + rowoff;
      float r = urand_(2u * (uint32_t)step, (uint32_t)row, (uint32_t)col);
      Hs[(size_t)row * HID + col] = (r < p) ? (_Float16)1.0f : (_Float16)0.0f;
    }
  }
}

// x = bernoulli(sigmoid(h W^T + bx)) -> Xh fp16 {0,1}.  K = H = 128.
// Wh B-tiles double-buffered through LDS (async DMA when available); output
// tiles transposed through LDS so each lane does one b128 store.
#define NT_CHUNK 4
#define CHUNK_HALVES (NT_CHUNK * 16 * HID)   // 8192 halves = 16 KB
__global__ void __launch_bounds__(128)
k_xstep(const _Float16* __restrict__ Hs, const _Float16* __restrict__ Wh,
        const float* __restrict__ bx, _Float16* __restrict__ Xh, int step) {
  __shared__ _Float16 sW[2][CHUNK_HALVES];   // 32 KB staged W
  __shared__ _Float16 sT[4][16 * 16];        // per-wave transpose tiles
  const int tid   = threadIdx.x;
  const int wid   = tid >> 5;
  const int lane  = tid & 31;
  const int wave  = blockIdx.x * 4 + wid;
  const int rbase = wave * 16;
  const int arow  = lane & 15;
  const int akoff = (lane >> 4) << 3;
  const int bcol  = lane & 15;
  const int bkoff = (lane >> 4) << 3;
  const int rowoff = (lane >> 4) << 3;

  // Load all 4 A-fragments (16x128 h tile) once; reuse across all 256 N tiles.
  F16x16 af[4];
  {
    const _Float16* ap = Hs + (size_t)(rbase + arow) * HID + akoff;
#pragma unroll
    for (int kc = 0; kc < 4; ++kc) {
      af[kc].h[0] = *(const h8*)(ap + kc * 32);
      af[kc].h[1] = *(const h8*)(ap + kc * 32 + 16);
    }
  }

  auto stage = [&](int c, int buf) {
    const _Float16* g = Wh + (size_t)c * CHUNK_HALVES;
    _Float16* l = &sW[buf][0];
#pragma unroll
    for (int i = 0; i < 8; ++i) {
      int off = (i * 128 + tid) * 8;   // 128 threads x 16B = 2KB contiguous per i
      copy16_g2l(g + off, l + off);
    }
  };

  const int NCHUNK = (DIM / 16) / NT_CHUNK;   // 64
  stage(0, 0);
  for (int c = 0; c < NCHUNK; ++c) {
    if (c + 1 < NCHUNK) {
      stage(c + 1, (c + 1) & 1);
      wait_async_le8();     // chunk c's (older) loads complete
    } else {
      wait_async_le0();
    }
    __syncthreads();        // all waves' chunk-c data visible in LDS
    const _Float16* wbuf = &sW[c & 1][0];
#pragma unroll
    for (int tl = 0; tl < NT_CHUNK; ++tl) {
      const int nt = c * NT_CHUNK + tl;
      v8f acc = {};
#pragma unroll
      for (int kc = 0; kc < 4; ++kc) {
        const _Float16* bp = wbuf + (size_t)(tl * 16 + bcol) * HID + kc * 32 + bkoff;
        F16x16 b;
        b.h[0] = *(const h8*)(bp);
        b.h[1] = *(const h8*)(bp + 16);
        acc = __builtin_amdgcn_wmma_f32_16x16x32_f16(
            false, af[kc].v, false, b.v, (short)0, acc, false, false);
      }
      const int colbase = nt * 16;
      const float bxc = bx[colbase + (lane & 15)];
      _Float16* tp = &sT[wid][0];
#pragma unroll
      for (int j = 0; j < 8; ++j) {
        float p = sigmoidf_(acc[j] + bxc);
        const int rl = j + rowoff;
        float r = urand_(2u * (uint32_t)step + 1u, (uint32_t)(rbase + rl),
                         (uint32_t)(colbase + (lane & 15)));
        tp[rl * 16 + (lane & 15)] = (r < p) ? (_Float16)1.0f : (_Float16)0.0f;
      }
      // Transposed writeback: lane -> (row = lane>>1, 8 cols), one b128 store.
      const int rl = lane >> 1;
      const int co = (lane & 1) * 8;
      h8 vrow = *(const h8*)(tp + rl * 16 + co);
      *(h8*)(Xh + (size_t)(rbase + rl) * DIM + colbase + co) = vrow;
    }
    __syncthreads();        // done reading sW[c&1] before it is overwritten
  }
}

__global__ void k_final(const float* __restrict__ acc, float* __restrict__ out) {
  out[0] = (acc[0] - acc[1]) * (1.0f / (float)BATCH);
}

// --- host ------------------------------------------------------------------

extern "C" void kernel_launch(void* const* d_in, const int* in_sizes, int n_in,
                              void* d_out, int out_size, void* d_ws, size_t ws_size,
                              hipStream_t stream) {
  (void)in_sizes; (void)n_in; (void)out_size; (void)ws_size;
  const float* x  = (const float*)d_in[0];
  const float* W  = (const float*)d_in[1];
  const float* bx = (const float*)d_in[2];
  const float* bh = (const float*)d_in[3];
  // d_in[4] = nb_gibbs_steps (==4), compiled in below.
  float* out = (float*)d_out;

  char* ws = (char*)d_ws;
  _Float16* Wt = (_Float16*)(ws);                      // 1.125 MB (padded)
  _Float16* Wh = (_Float16*)(ws + (2u << 20));         // 1 MB
  _Float16* Hs = (_Float16*)(ws + (3u << 20));         // 4 MB
  _Float16* Xh = (_Float16*)(ws + (8u << 20));         // 128 MB
  float*    acc = (float*)(ws + (136u << 20));         // 8 B

  k_init_w<<<DIM, HID, 0, stream>>>(W, bx, Wh, Wt, acc);
  k_x2half<<<(BATCH * (size_t)DIM / 4) / 256, 256, 0, stream>>>(x, Xh);

  const int gw = BATCH / (16 * 4);  // 256 blocks x 4 waves, one 16-row tile per wave

  // F(x) into acc[0]  (softplus + dot term fused in one WMMA pass)
  k_fe<<<gw, 128, 0, stream>>>(Xh, Wt, bh, acc + 0);

  for (int i = 0; i < 4; ++i) {   // CD-4 Gibbs chain
    k_hstep<<<gw, 128, 0, stream>>>(Xh, Wt, bh, Hs, i);
    k_xstep<<<gw, 128, 0, stream>>>(Hs, Wh, bx, Xh, i);
  }

  // F(x_rec) into acc[1]
  k_fe<<<gw, 128, 0, stream>>>(Xh, Wt, bh, acc + 1);

  k_final<<<1, 1, 0, stream>>>(acc, out);
}
